// GenericNC_39316130628017
// MI455X (gfx1250) — matrix operand order repaired and checked by reading, not compiled, not run on hardware
//
#include <hip/hip_runtime.h>
#include <hip/hip_bf16.h>

typedef __attribute__((ext_vector_type(16))) __bf16 v16bf;
typedef __attribute__((ext_vector_type(8)))  __bf16 v8bf;
typedef __attribute__((ext_vector_type(4)))  __bf16 v4bf;
typedef __attribute__((ext_vector_type(2)))  __bf16 v2bf;
typedef __attribute__((ext_vector_type(8)))  float  v8f;
typedef __attribute__((ext_vector_type(4)))  float  v4f;
typedef __attribute__((ext_vector_type(2)))  float  v2f;
typedef __attribute__((ext_vector_type(4)))  unsigned int v4u;
typedef __attribute__((ext_vector_type(8)))  int    v8i;
typedef __attribute__((ext_vector_type(4)))  int    v4i;

#define NB   16          // batch rows per workgroup (one WMMA M-tile)
#define D    512         // K of linear layer
#define NN   4096        // 64*64 output features
#define ZS   528         // bf16 stride per z row (1056 B, 16-B aligned)
#define XRS  66          // f32 stride per X_b row (8-B aligned frag loads)
#define XBS  (64 * XRS)  // 4224 floats per batch row of X
#define NSWEEP 10

// LDS partition (bytes)
#define OFF_SWF 16896                     // 16*ZS*2
#define OFF_SX  49664                     // OFF_SWF + 2 buffers * 8192 bf16
#define LDS_TOT (OFF_SX + NB * XBS * 4)   // 320000 B (~312.5 KB)

// ---- W staging helpers: global->VGPR (issued before WMMA) and
// ---- VGPR->LDS fragment store (issued after WMMA) for double buffering.
struct WStage { v4f r0[4]; v4f r1[4]; };

__device__ __forceinline__ void loadW(const float* __restrict__ W, int k0s,
                                      int c0, int tid, WStage& st) {
#pragma unroll
  for (int it = 0; it < 4; ++it) {
    const int i2  = tid + it * 256;
    const int a8  = i2 >> 5;              // 0..31, covers kk = 2*a8, 2*a8+1
    const int nn4 = (i2 & 31) << 2;       // 0..124
    const float* g0 = &W[(size_t)(k0s + 2 * a8) * NN + c0 + nn4];
    st.r0[it] = *(const v4f*)g0;
    st.r1[it] = *(const v4f*)(g0 + NN);
  }
}

__device__ __forceinline__ void storeW(__bf16* __restrict__ bufbase, int tid,
                                       const WStage& st) {
#pragma unroll
  for (int it = 0; it < 4; ++it) {
    const int i2  = tid + it * 256;
    const int a8  = i2 >> 5;
    const int nn4 = (i2 & 31) << 2;
    __bf16* fb = bufbase + ((a8 >> 4) << 12) + ((a8 & 7) << 1);
    const int lh16 = ((a8 >> 3) & 1) << 4;
#pragma unroll
    for (int u = 0; u < 4; ++u) {
      const int nn = nn4 + u;
      v2bf pk = { (__bf16)st.r0[it][u], (__bf16)st.r1[it][u] };
      *(v2bf*)(fb + ((nn >> 4) << 9) + ((lh16 + (nn & 15)) << 4)) = pk;
    }
  }
}

// ---------------------------------------------------------------------------
// Kernel 1: fused  X = z @ W + b  (bf16 WMMA, f32 acc)  then  G = X X^T
// (f32 WMMA).  One workgroup (8 waves) owns 16 batch rows; X stays in LDS.
// W is staged into LDS pre-swizzled into B-fragment order (contiguous 32-B
// fragment reads), double-buffered so global latency overlaps WMMA issue.
// ---------------------------------------------------------------------------
__global__ __launch_bounds__(256)
void fused_linear_gram(const float* __restrict__ z,
                       const float* __restrict__ W,
                       const float* __restrict__ bias,
                       float* __restrict__ gout) {
  extern __shared__ char smem[];
  __bf16* sZ  = (__bf16*)smem;                 // [16][ZS]   z chunk, bf16
  __bf16* sWf = (__bf16*)(smem + OFF_SWF);     // [2][2][8][32][16] B-fragments
  float*  sX  = (float*) (smem + OFF_SX);      // [16][64][XRS] X, padded

  const int tid  = threadIdx.x;
  const int lane = tid & 31;
  const int wv   = tid >> 5;       // wave 0..7
  const int m    = lane & 15;      // row/col within 16-tile
  const int hi   = lane >> 4;      // lane half-group

  // ---- stage z chunk as bf16 (float4 loads, 8B LDS stores) ----
  const size_t zbase = (size_t)blockIdx.x * NB * D;
  for (int i = tid; i < NB * D / 4; i += 256) {
    const int rr = i >> 7;               // 128 quads per 512-row
    const int cc = (i & 127) << 2;
    v4f q = *(const v4f*)&z[zbase + (rr << 9) + cc];
    v4bf pk = { (__bf16)q[0], (__bf16)q[1], (__bf16)q[2], (__bf16)q[3] };
    *(v4bf*)(sZ + rr * ZS + cc) = pk;
  }
  __syncthreads();

  // ---- GEMM: 32 column super-tiles of 128; wave wv owns columns wv*16.. ----
  for (int sup = 0; sup < NN / 128; ++sup) {
    const int c0 = sup * 128;
    v8f acc = {};
    WStage st;
    loadW(W, 0, c0, tid, st);            // prologue: fill buffer 0
    storeW(sWf, tid, st);
    for (int k0 = 0; k0 < D; k0 += 64) {
      __syncthreads();                   // buffer (k0>>6)&1 ready for all waves
      const int buf = (k0 >> 6) & 1;
      const bool have = (k0 + 64) < D;
      if (have) {
        if (k0 + 128 < D)
          __builtin_prefetch(&W[(size_t)(k0 + 128) * NN + c0 + ((tid & 7) << 4)], 0, 3);
        loadW(W, k0 + 64, c0, tid, st);  // global loads in flight during WMMA
      }
#pragma unroll
      for (int kb = 0; kb < 2; ++kb) {
        const __bf16* az = sZ + m * ZS + k0 + kb * 32 + 16 * hi;
        v8bf alo = *(const v8bf*)az;          // K = 16hi .. 16hi+7
        v8bf ahi = *(const v8bf*)(az + 8);    // K = 16hi+8 .. 16hi+15
        v16bf a = __builtin_shufflevector(alo, ahi, 0, 1, 2, 3, 4, 5, 6, 7,
                                          8, 9, 10, 11, 12, 13, 14, 15);
        v16bf b = *(const v16bf*)(sWf + (buf << 13) + (kb << 12) +
                                  (wv << 9) + (lane << 4));
        acc = __builtin_amdgcn_wmma_f32_16x16x32_bf16(false, a, false, b,
                                                      (short)0, acc, false, false);
      }
      if (have)
        storeW(sWf + ((buf ^ 1) << 13), tid, st);   // fill other buffer
    }
    // bias add + store tile into padded X (element [i][j] at 66*i + j)
    const int c  = c0 + wv * 16 + m;
    const float bc = bias[c];
    const int cp = c + 2 * (c >> 6);
#pragma unroll
    for (int r = 0; r < 8; ++r)
      sX[(r + 8 * hi) * XBS + cp] = acc[r] + bc;
  }
  __syncthreads();

  // ---- Gram: 16 batches x 16 (it,kt) tiles = 256 wave tasks, f32 WMMA ----
  for (int task = wv; task < NB * 16; task += 8) {
    const int bb = task >> 4;
    const int tt = task & 15;
    const int it = tt >> 2, kt = tt & 3;
    const float* Xb = sX + bb * XBS;
    v8f acc = {};
#pragma unroll 4
    for (int j0 = 0; j0 < 64; j0 += 4) {
      v2f a2 = *(const v2f*)&Xb[XRS * (it * 16 + m) + j0 + 2 * hi];
      v2f b2 = *(const v2f*)&Xb[XRS * (kt * 16 + m) + j0 + 2 * hi];
      acc = __builtin_amdgcn_wmma_f32_16x16x4_f32(false, a2, false, b2,
                                                  (short)0, acc, false, false);
    }
    const size_t obase = ((size_t)blockIdx.x * NB + bb) * (size_t)NN;
#pragma unroll
    for (int r = 0; r < 8; ++r)
      gout[obase + (size_t)(it * 16 + r + 8 * hi) * 64 + kt * 16 + m] = acc[r];
  }
}

// ---------------------------------------------------------------------------
// Kernel 2: batched 64x64 symmetric eigensolver (parallel cyclic Jacobi),
// one workgroup per matrix; reads G from d_out (via the Tensor Data Mover,
// padding rows 64->65 in flight), overwrites with sorted eigenvectors.
// ---------------------------------------------------------------------------
__global__ __launch_bounds__(256)
void jacobi_eigh(float* __restrict__ y) {
  __shared__ float A[64][65];
  __shared__ float V[64][65];
  __shared__ float rc[32], rs[32];
  __shared__ int   rp[32], rq[32];
  __shared__ float ev[64];
  __shared__ int   colsrc[64];

  const int tid = threadIdx.x;
  const size_t base = (size_t)blockIdx.x * 4096;

#if __has_builtin(__builtin_amdgcn_tensor_load_to_lds) && \
    __has_builtin(__builtin_amdgcn_s_wait_tensorcnt)
  // TDM: copy 4096 f32 G -> LDS, inserting 1-DWORD pad every 64 DWORDs
  // (pad_enable=1, pad_interval=5 -> 64 dw, pad_amount=0 -> 1 dw) so the
  // tile lands directly in the padded [64][65] layout.
  if (tid < 32) {
    const unsigned lds0 = (unsigned)(unsigned long long)(void*)&A[0][0];
    const unsigned long long ga = (unsigned long long)(const void*)(y + base);
    v4u g0 = { 1u,                                  // count=1, user descriptor
               lds0,                                // lds_addr
               (unsigned)(ga & 0xFFFFFFFFu),        // global_addr[31:0]
               (unsigned)(((ga >> 32) & 0x01FFFFFFu) | 0x80000000u) }; // type=2
    v8i g1 = { (int)0x01520000,   // data_size=4B, pad_en, interval=5, amount=0
               (int)0x10000000,   // tensor_dim0 = 4096 (bits 79:48 low half)
               (int)0x00010000,   // tensor_dim1 = 1
               (int)0x10000000,   // tile_dim0 = 4096
               1,                 // tile_dim1 = 1
               4096,              // tensor_dim0_stride = 4096
               (int)0x10000000,   // tensor_dim1_stride = 4096 (bits 223:208)
               0 };
    v4i gz4 = { 0, 0, 0, 0 };
    v8i gz8 = { 0, 0, 0, 0, 0, 0, 0, 0 };
    __builtin_amdgcn_tensor_load_to_lds(g0, g1, gz4, gz4, gz8, 0);
    __builtin_amdgcn_s_wait_tensorcnt(0);
  }
  for (int i = tid; i < 4096; i += 256)
    V[i >> 6][i & 63] = ((i >> 6) == (i & 63)) ? 1.f : 0.f;
#else
  for (int i = tid; i < 4096; i += 256) {
    A[i >> 6][i & 63] = y[base + i];
    V[i >> 6][i & 63] = ((i >> 6) == (i & 63)) ? 1.f : 0.f;
  }
#endif
  __syncthreads();

  for (int sweep = 0; sweep < NSWEEP; ++sweep) {
    for (int r = 0; r < 63; ++r) {
      // round-robin tournament: 32 disjoint (p,q) planes covering all 64 idx
      if (tid < 32) {
        int p, q;
        if (tid == 0) { p = 63; q = r; }
        else { p = (r + tid) % 63; q = (r + 63 - tid) % 63; }
        const float app = A[p][p], aqq = A[q][q], apq = A[p][q];
        float c = 1.f, s = 0.f;
        const float thresh = 1e-10f * (fabsf(app) + fabsf(aqq)) + 1e-35f;
        if (fabsf(apq) > thresh) {
          const float tau = (aqq - app) / (2.f * apq);
          const float t   = (tau >= 0.f ? 1.f : -1.f) /
                            (fabsf(tau) + sqrtf(1.f + tau * tau));
          c = 1.f / sqrtf(1.f + t * t);
          s = t * c;
        }
        rp[tid] = p; rq[tid] = q; rc[tid] = c; rs[tid] = s;
      }
      __syncthreads();
      // phase 1: column rotations  A <- A*J,  V <- V*J  (disjoint columns)
      for (int id = tid; id < 2048; id += 256) {
        const int pr = id >> 6, i = id & 63;
        const int p = rp[pr], q = rq[pr];
        const float c = rc[pr], s = rs[pr];
        const float aip = A[i][p], aiq = A[i][q];
        A[i][p] = c * aip - s * aiq;
        A[i][q] = s * aip + c * aiq;
        const float vip = V[i][p], viq = V[i][q];
        V[i][p] = c * vip - s * viq;
        V[i][q] = s * vip + c * viq;
      }
      __syncthreads();
      // phase 2: row rotations  A <- J^T*A  (disjoint rows)
      for (int id = tid; id < 2048; id += 256) {
        const int pr = id >> 6, j = id & 63;
        const int p = rp[pr], q = rq[pr];
        const float c = rc[pr], s = rs[pr];
        const float apj = A[p][j], aqj = A[q][j];
        A[p][j] = c * apj - s * aqj;
        A[q][j] = s * apj + c * aqj;
      }
      __syncthreads();
    }
  }

  if (tid < 64) ev[tid] = A[tid][tid];
  __syncthreads();
  if (tid < 64) {
    const float v = ev[tid];
    int rank = 0;
    for (int j = 0; j < 64; ++j) {
      const float u = ev[j];
      rank += (u < v) || (u == v && j < tid);
    }
    colsrc[rank] = tid;   // output column `rank` comes from V column `tid`
  }
  __syncthreads();
  for (int i = tid; i < 4096; i += 256)
    y[base + i] = V[i >> 6][colsrc[i & 63]];
}

// ---------------------------------------------------------------------------
extern "C" void kernel_launch(void* const* d_in, const int* in_sizes, int n_in,
                              void* d_out, int out_size, void* d_ws, size_t ws_size,
                              hipStream_t stream) {
  (void)n_in; (void)out_size; (void)d_ws; (void)ws_size;
  const float* z    = (const float*)d_in[0];
  const float* W    = (const float*)d_in[1];
  const float* bias = (const float*)d_in[2];
  float* y = (float*)d_out;

  const int B    = in_sizes[0] / D;   // 16384
  const int nblk = B / NB;            // 1024

  fused_linear_gram<<<nblk, 256, LDS_TOT, stream>>>(z, W, bias, y);
  jacobi_eigh<<<B, 256, 0, stream>>>(y);
}
